// Conv2dMem_24008867185117
// MI455X (gfx1250) — 8-bit matrix operand order repaired and checked by reading, compiled, not run on hardware
//
#include <hip/hip_runtime.h>


typedef __attribute__((ext_vector_type(8))) int   v8i;
typedef __attribute__((ext_vector_type(8))) float v8f;

#define BATCH 8
#define CIN   256
#define HW    56
#define NOUT  256
#define KDIM  2304
#define LDIM  3136
#define ROWS  (BATCH * LDIM)
#define NG    72
#define QMAX  63.0f
#define LT    32
#define NWAVE 8

__global__ void qweight_kernel(const float* __restrict__ w,
                               signed char* __restrict__ Bq8,
                               float* __restrict__ Sw) {
  const int g   = blockIdx.x >> 3;
  const int nb  = blockIdx.x & 7;
  const int tid = threadIdx.x;
  __shared__ float red[256];

  float vals[4];
  float m = 0.f;
#pragma unroll
  for (int j = 0; j < 4; ++j) {
    const int e  = tid + 256 * j;
    const int kk = e >> 5, nn = e & 31;
    const int n = nb * 32 + nn, k = g * 32 + kk;
    const float v = w[(size_t)n * KDIM + k];
    vals[j] = v;
    m = fmaxf(m, fabsf(v));
  }
  red[tid] = m;
  __syncthreads();
  for (int s = 128; s > 0; s >>= 1) {
    if (tid < s) red[tid] = fmaxf(red[tid], red[tid + s]);
    __syncthreads();
  }
  const float ma    = red[0];
  const float scale = (ma == 0.f) ? 1.f : ma * (1.0f / QMAX);
#pragma unroll
  for (int j = 0; j < 4; ++j) {
    const int e  = tid + 256 * j;
    const int kk = e >> 5, nn = e & 31;
    const int n = nb * 32 + nn, k = g * 32 + kk;
    const float q = fminf(QMAX, fmaxf(-QMAX, rintf(vals[j] / scale)));
    Bq8[(size_t)n * KDIM + k] = (signed char)(int)q;
  }
  if (tid == 0) Sw[blockIdx.x] = scale;
}

__global__ void __launch_bounds__(256) conv_kernel(
    const float* __restrict__ x, const signed char* __restrict__ Bq8,
    const float* __restrict__ Sw, const float* __restrict__ bias,
    float* __restrict__ out) {
  __shared__ signed char Al[LT * KDIM];
  __shared__ float       Sxl[NG * LT];

  const int tid  = threadIdx.x;
  const int w    = tid >> 5;
  const int lane = tid & 31;
  const int r0   = blockIdx.x * LT;
  const int b    = r0 / LDIM;
  const int lbase = r0 - b * LDIM;

  {
    const int gs = lane >> 3;
    const int s  = lane & 7;
    for (int j = w; j < LT * (NG / 4); j += NWAVE) {
      const int rr = j / (NG / 4);
      const int g4 = j - rr * (NG / 4);
      const int g  = g4 * 4 + gs;
      const int l  = lbase + rr;
      const int ho = l / HW, wo = l - ho * HW;
      const int kb = g * 32 + s * 4;

      float v[4];
      float m = 0.f;
#pragma unroll
      for (int e = 0; e < 4; ++e) {
        const int k   = kb + e;
        const int c   = k / 9;
        const int rem = k - c * 9;
        const int ki  = rem / 3;
        const int kj  = rem - ki * 3;
        const int hin = ho + ki - 1, win = wo + kj - 1;
        float vv = 0.f;
        if ((unsigned)hin < HW && (unsigned)win < HW)
          vv = x[((b * CIN + c) * HW + hin) * HW + win];
        v[e] = vv;
        m = fmaxf(m, fabsf(vv));
      }
#pragma unroll
      for (int off = 1; off < 8; off <<= 1)
        m = fmaxf(m, __shfl_xor(m, off, 32));

      const float inv = (m == 0.f) ? 0.f : QMAX / m;
      int p = 0;
#pragma unroll
      for (int e = 0; e < 4; ++e) {
        const int q = (int)fminf(QMAX, fmaxf(-QMAX, rintf(v[e] * inv)));
        p |= (q & 0xff) << (8 * e);
      }
      ((int*)Al)[rr * (KDIM / 4) + (kb >> 2)] = p;
      if (s == 0) Sxl[g * LT + rr] = (m == 0.f) ? 1.f : m * (1.0f / QMAX);
    }
  }
  __syncthreads();

  const int hi = lane >> 4;
  const int ml = lane & 15;
  const int nwb = w * 32;

  v8f acc[4];
#pragma unroll
  for (int t = 0; t < 4; ++t) acc[t] = {};

#pragma unroll 1
  for (int g = 0; g < NG; ++g) {
    const int kg = g * 32;
    const float sw = Sw[g * 8 + w];

    v8i bmat[2];
#pragma unroll
    for (int ns = 0; ns < 2; ++ns) {
      const int n = nwb + ns * 16 + ml;
      const int* bp = (const int*)(Bq8 + (size_t)n * KDIM + kg + hi * 16);
      if (g + 1 < NG) __builtin_prefetch(bp + 8, 0, 0);
      v8i bm = {bp[0], bp[1], bp[2], bp[3], 0, 0, 0, 0};
      bmat[ns] = bm;
    }

#pragma unroll
    for (int as = 0; as < 2; ++as) {
      const int rowA = as * 16 + ml;
      const int* ap = (const int*)(Al + rowA * KDIM + kg);
      const int h2 = hi * 4;
      v8i amat = {ap[h2], ap[h2 + 1], ap[h2 + 2], ap[h2 + 3], 0, 0, 0, 0};

      const v8f sx8 = *(const v8f*)&Sxl[g * LT + as * 16 + hi * 8];
      float f[8];
#pragma unroll
      for (int i = 0; i < 8; ++i) f[i] = sx8[i] * sw;

#pragma unroll
      for (int ns = 0; ns < 2; ++ns) {
        v8i cz = {};
        v8i d = __builtin_amdgcn_wmma_i32_16x16x64_iu8(
            true, amat, true, bmat[ns], cz, false, false);
#pragma unroll
        for (int i = 0; i < 8; ++i)
          acc[as * 2 + ns][i] += (float)d[i] * f[i];
      }
    }
  }

#pragma unroll
  for (int as = 0; as < 2; ++as) {
#pragma unroll
    for (int ns = 0; ns < 2; ++ns) {
      const int n = nwb + ns * 16 + ml;
      const float bb = bias[n];
      v8f res;
#pragma unroll
      for (int i = 0; i < 8; ++i) res[i] = acc[as * 2 + ns][i] + bb;
      float* op = out + (size_t)(b * NOUT + n) * LDIM + lbase + as * 16 + hi * 8;
      *(v8f*)op = res;
    }
  }
}

extern "C" void kernel_launch(void* const* d_in, const int* in_sizes, int n_in,
                              void* d_out, int out_size, void* d_ws, size_t ws_size,
                              hipStream_t stream) {
  (void)in_sizes; (void)n_in; (void)out_size; (void)ws_size;
  const float* x    = (const float*)d_in[0];
  const float* w    = (const float*)d_in[1];
  const float* bias = (const float*)d_in[2];
  float* out = (float*)d_out;

  char* ws = (char*)d_ws;
  signed char* Bq8 = (signed char*)ws;
  float*       Sw  = (float*)(ws + (size_t)KDIM * NOUT);

  qweight_kernel<<<NG * 8, 256, 0, stream>>>(w, Bq8, Sw);
  conv_kernel<<<ROWS / LT, 256, 0, stream>>>(x, Bq8, Sw, bias, out);
}
